// DTFD_MIL_19791209300143
// MI455X (gfx1250) — compile-verified, matrix operand order checked by reading, not run on hardware
//
#include <hip/hip_runtime.h>
#include <cstddef>
#include <cstdint>

// ---------------------------------------------------------------------------
// DTFD-MIL fused pipeline for MI455X (gfx1250, wave32, WMMA).
//
// Roofline: x is 8*16384*512 f32 = 256 MB -> ~11 us at 23.3 TB/s. We make a
// single pass over x (fused GEMM + gated attention + online softmax partials)
// so HBM traffic ~= that floor. Weights (<3 MB) stay L2-resident (192 MB L2).
// Matrix math uses V_WMMA_F32_16X16X32_BF16 (f32 accumulate).
// ---------------------------------------------------------------------------

#define B_     8
#define NINST  16384
#define D_     512
#define DH_    256          // D/2
#define C_     2
#define G_     8
#define K_     4
#define BUF_   1024
#define DA_    128
#define M_     (NINST / G_) // 2048 instances per (bag, group)
#define NNODES (B_ + BUF_)  // 1032
#define ROWS   64           // rows per main-kernel block
#define NBLK   (M_ / ROWS)  // 32 blocks per group
#define NGROUP (B_ * G_)    // 64
#define STRIDE 528          // LDS bf16 row stride (row byte size 1056 = 33*32)

// workspace layout (bytes)
#define OFF_PWDR 0u                         // 512*512 bf16 fragments
#define OFF_PWV1 (OFF_PWDR + 524288u)       // 512*128 bf16
#define OFF_PWU1 (OFF_PWV1 + 131072u)       // 512*128 bf16
#define OFF_PART (OFF_PWU1 + 131072u)       // 64*32*514 f32 online-softmax partials
#define OFF_AF   (OFF_PART + 4210688u)      // 64*512 f32
#define OFF_BAGF (OFF_AF   + 131072u)       // 8*512 f32
#define OFF_XCAT (OFF_BAGF + 16384u)        // 1032*512 f32
#define OFF_XN   (OFF_XCAT + 2113536u)      // 1032*256 f32
#define OFF_IDX  (OFF_XN   + 1056768u)      // 1032*4 int
#define OFF_H    (OFF_IDX  + 16896u)        // 1032*512 f32
#define OFF_H1   (OFF_H    + 2113536u)      // 1032*512 f32

#define SMEM_MAIN (2 * ROWS * STRIDE * 2 + 512)   // xs + ms (bf16) + ared

typedef __attribute__((ext_vector_type(16))) __bf16         v16bf;
typedef __attribute__((ext_vector_type(16))) unsigned short v16u;
typedef __attribute__((ext_vector_type(8)))  float          v8f;

union Frag { v16u u; v16bf b; };

static __device__ __forceinline__ unsigned short f2bf(float f) {
  unsigned u = __float_as_uint(f);
  u += 0x7FFFu + ((u >> 16) & 1u);           // round-to-nearest-even
  return (unsigned short)(u >> 16);
}
static __device__ __forceinline__ float bf2f(unsigned short h) {
  return __uint_as_float(((unsigned)h) << 16);
}
static __device__ __forceinline__ float sigm(float x) {
  return 1.0f / (1.0f + __expf(-x));
}

// ---------------------------------------------------------------------------
// Kernel 0: convert Wdr/Wv1/Wu1 f32 -> bf16, pre-swizzled into per-lane
// B-fragments: for tile (nt,kt), lane l covers column N = nt*16 + (l&15),
// K = kt*32 + (l>>4)*16 + e, e = 0..15 contiguous => one 32B load per frag.
// ---------------------------------------------------------------------------
__global__ void __launch_bounds__(256)
dtfd_pack_weights(const float* __restrict__ Wdr, const float* __restrict__ Wv1,
                  const float* __restrict__ Wu1,
                  unsigned short* __restrict__ pWdr,
                  unsigned short* __restrict__ pWv1,
                  unsigned short* __restrict__ pWu1)
{
  const int tid = blockIdx.x * 256 + threadIdx.x;   // 1536*256 = 393216 exact
  const float* src; unsigned short* dst; int ncols; int f;
  if (tid < 262144)      { src = Wdr; dst = pWdr; ncols = 512; f = tid; }
  else if (tid < 327680) { src = Wv1; dst = pWv1; ncols = 128; f = tid - 262144; }
  else                   { src = Wu1; dst = pWu1; ncols = 128; f = tid - 327680; }
  const int e  = f & 15;
  const int l  = (f >> 4) & 31;
  const int nk = f >> 9;             // nt*16 + kt
  const int kt = nk & 15;
  const int nt = nk >> 4;
  const int k  = kt * 32 + ((l >> 4) << 4) + e;
  const int n  = nt * 16 + (l & 15);
  dst[f] = f2bf(src[(size_t)k * ncols + n]);
}

// ---------------------------------------------------------------------------
// Kernel 1 (dominant): per 64-row tile of one (b,g) group:
//   mid = relu(x @ Wdr + bdr)                 (WMMA bf16, LDS resident)
//   V = tanh(mid@Wv1+bv1), U = sigm(mid@Wu1+bu1)   (WMMA bf16)
//   a = (V*U)@Ww1 + bw1 ; online-softmax block partial of sum_r e^a_r mid_r
// ---------------------------------------------------------------------------
__global__ void __launch_bounds__(256)
dtfd_main_fused(const float* __restrict__ x,
                const float* __restrict__ bdr,
                const float* __restrict__ bv1,
                const float* __restrict__ bu1,
                const float* __restrict__ Ww1,
                const float* __restrict__ bw1,
                const unsigned short* __restrict__ pWdr,
                const unsigned short* __restrict__ pWv1,
                const unsigned short* __restrict__ pWu1,
                float* __restrict__ partials)
{
  extern __shared__ char smem[];
  unsigned short* xs  = (unsigned short*)smem;                      // 64x528 bf16
  unsigned short* ms  = (unsigned short*)(smem + ROWS * STRIDE * 2);// 64x528 bf16
  float*          vu  = (float*)smem;                // aliases xs after GEMM1
  float*          ared = (float*)(smem + 2 * ROWS * STRIDE * 2);    // 66 floats

  const int tid  = threadIdx.x;
  const int lane = tid & 31;
  const int wave = tid >> 5;

  const int gid = blockIdx.x >> 5;   // 0..63  == b*G + g
  const int blk = blockIdx.x & 31;
  const int b   = gid >> 3;
  const int g   = gid & 7;
  const size_t xbase = ((size_t)b * NINST + (size_t)g * M_ + (size_t)blk * ROWS) * D_;

  // stage x tile: f32 global -> bf16 LDS (single pass over x per element)
  for (int i = tid; i < ROWS * D_; i += 256) {
    const int r = i >> 9, c = i & (D_ - 1);
    xs[r * STRIDE + c] = f2bf(x[xbase + (size_t)r * D_ + c]);
  }
  __syncthreads();

  // ---- GEMM1: mid = relu(x @ Wdr + bdr). Wave w owns N-tiles [4w,4w+4). ----
  for (int nti = 0; nti < 4; ++nti) {
    const int nt = wave * 4 + nti;                    // 0..31
    v8f acc[4] = {};
    for (int kt = 0; kt < 16; ++kt) {
      Frag bf_;
      bf_.u = *(const v16u*)&pWdr[(((size_t)nt * 16 + kt) * 32 + lane) * 16];
#pragma unroll
      for (int mt = 0; mt < 4; ++mt) {
        Frag af_;
        const int row = mt * 16 + (lane & 15);
        const int col = kt * 32 + ((lane >> 4) << 4);
        af_.u = *(const v16u*)&xs[row * STRIDE + col];
        acc[mt] = __builtin_amdgcn_wmma_f32_16x16x32_bf16(
            false, af_.b, false, bf_.b, (short)0, acc[mt], false, false);
      }
    }
    const int cn   = nt * 16 + (lane & 15);
    const float bi = bdr[cn];
#pragma unroll
    for (int mt = 0; mt < 4; ++mt) {
#pragma unroll
      for (int j = 0; j < 8; ++j) {
        const int row = mt * 16 + ((lane < 16) ? j : j + 8);  // ISA C/D layout
        float v = acc[mt][j] + bi;
        ms[row * STRIDE + cn] = f2bf(v > 0.f ? v : 0.f);
      }
    }
  }
  __syncthreads();

  // ---- GEMM2: waves 0-3 -> V = tanh(mid@Wv1+bv1); waves 4-7 -> U sigmoid. ----
  {
    const unsigned short* pW2 = (wave < 4) ? pWv1 : pWu1;
    const float*          b2  = (wave < 4) ? bv1 : bu1;
    const int w2 = wave & 3;
    v8f acc[2][4] = {};
    for (int ni = 0; ni < 2; ++ni) {
      const int nt = w2 * 2 + ni;                     // 0..7
      for (int kt = 0; kt < 16; ++kt) {
        Frag bf_;
        bf_.u = *(const v16u*)&pW2[(((size_t)nt * 16 + kt) * 32 + lane) * 16];
#pragma unroll
        for (int mt = 0; mt < 4; ++mt) {
          Frag af_;
          const int row = mt * 16 + (lane & 15);
          const int col = kt * 32 + ((lane >> 4) << 4);
          af_.u = *(const v16u*)&ms[row * STRIDE + col];
          acc[ni][mt] = __builtin_amdgcn_wmma_f32_16x16x32_bf16(
              false, af_.b, false, bf_.b, (short)0, acc[ni][mt], false, false);
        }
      }
    }
    if (wave < 4) {                                   // V -> vu (f32, aliases xs)
#pragma unroll
      for (int ni = 0; ni < 2; ++ni) {
        const int cn = (w2 * 2 + ni) * 16 + (lane & 15);
        const float bi = b2[cn];
#pragma unroll
        for (int mt = 0; mt < 4; ++mt)
#pragma unroll
          for (int j = 0; j < 8; ++j) {
            const int row = mt * 16 + ((lane < 16) ? j : j + 8);
            vu[row * DA_ + cn] = tanhf(acc[ni][mt][j] + bi);
          }
      }
    }
    __syncthreads();
    if (wave >= 4) {                                  // vu *= sigmoid(U)
#pragma unroll
      for (int ni = 0; ni < 2; ++ni) {
        const int cn = (w2 * 2 + ni) * 16 + (lane & 15);
        const float bi = b2[cn];
#pragma unroll
        for (int mt = 0; mt < 4; ++mt)
#pragma unroll
          for (int j = 0; j < 8; ++j) {
            const int row = mt * 16 + ((lane < 16) ? j : j + 8);
            vu[row * DA_ + cn] *= sigm(acc[ni][mt][j] + bi);
          }
      }
    }
    __syncthreads();
  }

  // ---- a[r] = vu[r,:] . Ww1 + bw1 ; block-local softmax partials ----
  if (tid < ROWS) {
    float s = bw1[0];
    for (int c = 0; c < DA_; ++c) s += vu[tid * DA_ + c] * Ww1[c];
    ared[tid] = s;
  }
  __syncthreads();
  if (tid == 0) {
    float m = ared[0];
    for (int r = 1; r < ROWS; ++r) m = fmaxf(m, ared[r]);
    float ssum = 0.f;
    for (int r = 0; r < ROWS; ++r) { float e = __expf(ared[r] - m); ared[r] = e; ssum += e; }
    ared[ROWS] = m; ared[ROWS + 1] = ssum;
  }
  __syncthreads();
  float* part = partials + ((size_t)gid * NBLK + blk) * 514;
  for (int c = tid; c < D_; c += 256) {
    float a = 0.f;
    for (int r = 0; r < ROWS; ++r) a += ared[r] * bf2f(ms[r * STRIDE + c]);
    part[c] = a;
  }
  if (tid == 0) { part[512] = ared[ROWS]; part[513] = ared[ROWS + 1]; }
}

// ---------------------------------------------------------------------------
// Kernel 2: combine 32 online-softmax partials per group -> af; sub_preds.
// ---------------------------------------------------------------------------
__global__ void __launch_bounds__(256)
dtfd_reduce_af(const float* __restrict__ partials,
               const float* __restrict__ Wsc, const float* __restrict__ bsc,
               float* __restrict__ af, float* __restrict__ out)
{
  __shared__ float wgt[NBLK];
  __shared__ float afl[D_];
  __shared__ float r0[256], r1[256];
  const int gid = blockIdx.x;
  const float* base = partials + (size_t)gid * NBLK * 514;
  if (threadIdx.x == 0) {
    float M = -3.0e38f;
    for (int i = 0; i < NBLK; ++i) M = fmaxf(M, base[i * 514 + 512]);
    float S = 0.f;
    for (int i = 0; i < NBLK; ++i) {
      float e = __expf(base[i * 514 + 512] - M);
      wgt[i] = e;
      S += e * base[i * 514 + 513];
    }
    const float inv = 1.f / S;
    for (int i = 0; i < NBLK; ++i) wgt[i] *= inv;
  }
  __syncthreads();
  for (int c = threadIdx.x; c < D_; c += 256) {
    float a = 0.f;
    for (int i = 0; i < NBLK; ++i) a += wgt[i] * base[i * 514 + c];
    afl[c] = a;
    af[(size_t)gid * D_ + c] = a;
  }
  __syncthreads();
  float p0 = 0.f, p1 = 0.f;
  for (int d = threadIdx.x; d < D_; d += 256) {
    p0 += afl[d] * Wsc[d * 2 + 0];
    p1 += afl[d] * Wsc[d * 2 + 1];
  }
  r0[threadIdx.x] = p0; r1[threadIdx.x] = p1;
  __syncthreads();
  for (int s = 128; s > 0; s >>= 1) {
    if (threadIdx.x < s) { r0[threadIdx.x] += r0[threadIdx.x + s]; r1[threadIdx.x] += r1[threadIdx.x + s]; }
    __syncthreads();
  }
  if (threadIdx.x == 0) {
    const int bb = gid >> 3, gg = gid & 7;       // gid = b*G+g ; cat row = g*B+b
    out[16 + (gg * B_ + bb) * 2 + 0] = r0[0] + bsc[0];
    out[16 + (gg * B_ + bb) * 2 + 1] = r1[0] + bsc[1];
  }
}

// ---------------------------------------------------------------------------
// Kernel 3: second gated attention over groups -> bag_feat, bag_pred.
// Tiny (64x512 input); single block, plain VALU.
// ---------------------------------------------------------------------------
__global__ void __launch_bounds__(256)
dtfd_attn2(const float* __restrict__ af,
           const float* __restrict__ Wv2, const float* __restrict__ bv2,
           const float* __restrict__ Wu2, const float* __restrict__ bu2,
           const float* __restrict__ Ww2, const float* __restrict__ bw2,
           const float* __restrict__ Wbc, const float* __restrict__ bbc,
           float* __restrict__ bag_feat, float* __restrict__ x_concat,
           float* __restrict__ out)
{
  __shared__ float a2p[256];
  __shared__ float A2[NGROUP];
  __shared__ float bagF[B_ * D_];
  const int t = threadIdx.x;
  const int r = t & 63;            // row = b*G+g
  const int chunk = t >> 6;        // 4 column chunks of 32
  float partial = 0.f;
  for (int ci = 0; ci < 32; ++ci) {
    const int c = chunk * 32 + ci;
    float v = bv2[c], u = bu2[c];
    for (int d = 0; d < D_; ++d) {
      const float a = af[(size_t)r * D_ + d];
      v += a * Wv2[d * DA_ + c];
      u += a * Wu2[d * DA_ + c];
    }
    partial += tanhf(v) * sigm(u) * Ww2[c];
  }
  a2p[t] = partial;
  __syncthreads();
  if (t < NGROUP) A2[t] = a2p[t] + a2p[t + 64] + a2p[t + 128] + a2p[t + 192] + bw2[0];
  __syncthreads();
  if (t < B_) {                    // softmax over g for bag t
    float m = A2[t * G_];
    for (int gg = 1; gg < G_; ++gg) m = fmaxf(m, A2[t * G_ + gg]);
    float s = 0.f;
    for (int gg = 0; gg < G_; ++gg) { float e = __expf(A2[t * G_ + gg] - m); A2[t * G_ + gg] = e; s += e; }
    const float inv = 1.f / s;
    for (int gg = 0; gg < G_; ++gg) A2[t * G_ + gg] *= inv;
  }
  __syncthreads();
  for (int i = t; i < B_ * D_; i += 256) {
    const int bb = i >> 9, d = i & (D_ - 1);
    float acc = 0.f;
    for (int gg = 0; gg < G_; ++gg) acc += A2[bb * G_ + gg] * af[(size_t)(bb * G_ + gg) * D_ + d];
    bagF[i] = acc;
    bag_feat[i] = acc;
    x_concat[i] = acc;             // rows 0..7 of x_concat
  }
  __syncthreads();
  if (t < 16) {                    // bag_pred -> out[0..15]
    const int bb = t >> 1, c = t & 1;
    float acc = bbc[c];
    for (int d = 0; d < D_; ++d) acc += bagF[bb * D_ + d] * Wbc[d * 2 + c];
    out[bb * 2 + c] = acc;
  }
}

__global__ void __launch_bounds__(256)
dtfd_copy_rehearsal(const float* __restrict__ rehearsal, float* __restrict__ x_concat)
{
  const size_t i = (size_t)blockIdx.x * 256 + threadIdx.x;   // 2048*256 == BUF_*D_
  x_concat[(size_t)B_ * D_ + i] = rehearsal[i];
}

// Kernel 4: _x = leaky_relu(x_concat @ Wf + bf); xn = _x / (||_x|| + 1e-12)
__global__ void __launch_bounds__(256)
dtfd_feat_norm(const float* __restrict__ x_concat, const float* __restrict__ Wf,
               const float* __restrict__ bf_, float* __restrict__ xn)
{
  __shared__ float xrow[D_];
  __shared__ float red[256];
  const int i = blockIdx.x;
  const int c = threadIdx.x;       // DH_ == 256 output columns
  for (int d = threadIdx.x; d < D_; d += 256) xrow[d] = x_concat[(size_t)i * D_ + d];
  __syncthreads();
  float acc = bf_[c];
  for (int d = 0; d < D_; ++d) acc += xrow[d] * Wf[d * DH_ + c];
  acc = (acc > 0.f) ? acc : 0.01f * acc;
  red[c] = acc * acc;
  __syncthreads();
  for (int s = 128; s > 0; s >>= 1) { if (c < s) red[c] += red[c + s]; __syncthreads(); }
  const float nrm = sqrtf(red[0]) + 1e-12f;
  xn[(size_t)i * DH_ + c] = acc / nrm;
}

// Kernel 5: cosine sim row + stable top-4 (ties -> lowest index)
__global__ void __launch_bounds__(256)
dtfd_topk(const float* __restrict__ xn, int* __restrict__ idx)
{
  __shared__ float xi[DH_];
  __shared__ float cv[256 * 4];
  __shared__ int   ci[256 * 4];
  const int i = blockIdx.x;
  xi[threadIdx.x] = xn[(size_t)i * DH_ + threadIdx.x];
  __syncthreads();
  float bv[4] = {-3e38f, -3e38f, -3e38f, -3e38f};
  int   bi[4] = {0x7fffffff, 0x7fffffff, 0x7fffffff, 0x7fffffff};
  for (int j = threadIdx.x; j < NNODES; j += 256) {
    const float* xj = xn + (size_t)j * DH_;
    float dp = 0.f;
    for (int c = 0; c < DH_; ++c) dp += xi[c] * xj[c];
    for (int k = 0; k < 4; ++k) {
      if (dp > bv[k] || (dp == bv[k] && j < bi[k])) {
        for (int q = 3; q > k; --q) { bv[q] = bv[q - 1]; bi[q] = bi[q - 1]; }
        bv[k] = dp; bi[k] = j;
        break;
      }
    }
  }
  for (int k = 0; k < 4; ++k) { cv[threadIdx.x * 4 + k] = bv[k]; ci[threadIdx.x * 4 + k] = bi[k]; }
  __syncthreads();
  if (threadIdx.x == 0) {
    for (int k = 0; k < 4; ++k) {
      float best = -3e38f; int besti = 0x7fffffff; int slot = 0;
      for (int s = 0; s < 256 * 4; ++s) {
        if (ci[s] < 0) continue;
        if (cv[s] > best || (cv[s] == best && ci[s] < besti)) { best = cv[s]; besti = ci[s]; slot = s; }
      }
      idx[i * 4 + k] = besti;
      ci[slot] = -1;
    }
  }
}

// Kernel 6: h = pad(bag_feat) + mean_k x_concat[idx]
__global__ void __launch_bounds__(256)
dtfd_gather_h(const float* __restrict__ x_concat, const int* __restrict__ idx,
              const float* __restrict__ bag_feat, float* __restrict__ h)
{
  const int i = blockIdx.x;
  const int i0 = idx[i * 4 + 0], i1 = idx[i * 4 + 1], i2 = idx[i * 4 + 2], i3 = idx[i * 4 + 3];
  for (int d = threadIdx.x; d < D_; d += 256) {
    const float e = 0.25f * (x_concat[(size_t)i0 * D_ + d] + x_concat[(size_t)i1 * D_ + d] +
                             x_concat[(size_t)i2 * D_ + d] + x_concat[(size_t)i3 * D_ + d]);
    const float p = (i < B_) ? bag_feat[i * D_ + d] : 0.f;
    h[(size_t)i * D_ + d] = p + e;
  }
}

// Kernel 7: h1 = relu(mean_k h[idx] @ Wg1 + h @ Wg1s + bg1)
__global__ void __launch_bounds__(256)
dtfd_gcn1(const float* __restrict__ h, const int* __restrict__ idx,
          const float* __restrict__ Wg1, const float* __restrict__ Wg1s,
          const float* __restrict__ bg1, float* __restrict__ h1)
{
  __shared__ float m1[D_], hr[D_];
  const int i = blockIdx.x;
  const int i0 = idx[i * 4 + 0], i1 = idx[i * 4 + 1], i2 = idx[i * 4 + 2], i3 = idx[i * 4 + 3];
  for (int d = threadIdx.x; d < D_; d += 256) {
    m1[d] = 0.25f * (h[(size_t)i0 * D_ + d] + h[(size_t)i1 * D_ + d] +
                     h[(size_t)i2 * D_ + d] + h[(size_t)i3 * D_ + d]);
    hr[d] = h[(size_t)i * D_ + d];
  }
  __syncthreads();
  for (int c = threadIdx.x; c < D_; c += 256) {
    float acc = bg1[c];
    for (int d = 0; d < D_; ++d) acc += m1[d] * Wg1[d * D_ + c] + hr[d] * Wg1s[d * D_ + c];
    h1[(size_t)i * D_ + c] = fmaxf(acc, 0.f);
  }
}

// Kernel 8: logits[:B] = mean_k h1[idx] @ Wg2 + h1 @ Wg2s + bg2
__global__ void __launch_bounds__(256)
dtfd_gcn2(const float* __restrict__ h1, const int* __restrict__ idx,
          const float* __restrict__ Wg2, const float* __restrict__ Wg2s,
          const float* __restrict__ bg2, float* __restrict__ out)
{
  __shared__ float m2[D_], hr[D_];
  __shared__ float r0[256], r1[256];
  const int i = blockIdx.x;        // 0..7
  const int i0 = idx[i * 4 + 0], i1 = idx[i * 4 + 1], i2 = idx[i * 4 + 2], i3 = idx[i * 4 + 3];
  for (int d = threadIdx.x; d < D_; d += 256) {
    m2[d] = 0.25f * (h1[(size_t)i0 * D_ + d] + h1[(size_t)i1 * D_ + d] +
                     h1[(size_t)i2 * D_ + d] + h1[(size_t)i3 * D_ + d]);
    hr[d] = h1[(size_t)i * D_ + d];
  }
  __syncthreads();
  float p0 = 0.f, p1 = 0.f;
  for (int d = threadIdx.x; d < D_; d += 256) {
    p0 += m2[d] * Wg2[d * 2 + 0] + hr[d] * Wg2s[d * 2 + 0];
    p1 += m2[d] * Wg2[d * 2 + 1] + hr[d] * Wg2s[d * 2 + 1];
  }
  r0[threadIdx.x] = p0; r1[threadIdx.x] = p1;
  __syncthreads();
  for (int s = 128; s > 0; s >>= 1) {
    if (threadIdx.x < s) { r0[threadIdx.x] += r0[threadIdx.x + s]; r1[threadIdx.x] += r1[threadIdx.x + s]; }
    __syncthreads();
  }
  if (threadIdx.x == 0) {
    out[144 + i * 2 + 0] = r0[0] + bg2[0];
    out[144 + i * 2 + 1] = r1[0] + bg2[1];
  }
}

// ---------------------------------------------------------------------------
extern "C" void kernel_launch(void* const* d_in, const int* in_sizes, int n_in,
                              void* d_out, int out_size, void* d_ws, size_t ws_size,
                              hipStream_t stream)
{
  (void)in_sizes; (void)n_in; (void)out_size; (void)ws_size;
  const float* x         = (const float*)d_in[0];
  const float* rehearsal = (const float*)d_in[1];
  const float* Wdr  = (const float*)d_in[2];
  const float* bdr  = (const float*)d_in[3];
  const float* Wv1  = (const float*)d_in[4];
  const float* bv1  = (const float*)d_in[5];
  const float* Wu1  = (const float*)d_in[6];
  const float* bu1  = (const float*)d_in[7];
  const float* Ww1  = (const float*)d_in[8];
  const float* bw1  = (const float*)d_in[9];
  const float* Wsc  = (const float*)d_in[10];
  const float* bsc  = (const float*)d_in[11];
  const float* Wv2  = (const float*)d_in[12];
  const float* bv2  = (const float*)d_in[13];
  const float* Wu2  = (const float*)d_in[14];
  const float* bu2  = (const float*)d_in[15];
  const float* Ww2  = (const float*)d_in[16];
  const float* bw2  = (const float*)d_in[17];
  const float* Wbc  = (const float*)d_in[18];
  const float* bbc  = (const float*)d_in[19];
  const float* Wf   = (const float*)d_in[20];
  const float* bf_  = (const float*)d_in[21];
  const float* Wg1  = (const float*)d_in[22];
  const float* Wg1s = (const float*)d_in[23];
  const float* bg1  = (const float*)d_in[24];
  const float* Wg2  = (const float*)d_in[25];
  const float* Wg2s = (const float*)d_in[26];
  const float* bg2  = (const float*)d_in[27];
  float* out = (float*)d_out;

  char* ws = (char*)d_ws;
  unsigned short* pWdr = (unsigned short*)(ws + OFF_PWDR);
  unsigned short* pWv1 = (unsigned short*)(ws + OFF_PWV1);
  unsigned short* pWu1 = (unsigned short*)(ws + OFF_PWU1);
  float* partials = (float*)(ws + OFF_PART);
  float* af       = (float*)(ws + OFF_AF);
  float* bag_feat = (float*)(ws + OFF_BAGF);
  float* x_concat = (float*)(ws + OFF_XCAT);
  float* xn       = (float*)(ws + OFF_XN);
  int*   idx      = (int*)  (ws + OFF_IDX);
  float* h        = (float*)(ws + OFF_H);
  float* h1       = (float*)(ws + OFF_H1);

  // main kernel needs ~133 KB dynamic LDS (well under the 320 KB WGP pool)
  (void)hipFuncSetAttribute((const void*)dtfd_main_fused,
                            hipFuncAttributeMaxDynamicSharedMemorySize, 147456);

  dtfd_pack_weights<<<1536, 256, 0, stream>>>(Wdr, Wv1, Wu1, pWdr, pWv1, pWu1);
  dtfd_main_fused<<<NGROUP * NBLK, 256, SMEM_MAIN, stream>>>(
      x, bdr, bv1, bu1, Ww1, bw1, pWdr, pWv1, pWu1, partials);
  dtfd_reduce_af<<<NGROUP, 256, 0, stream>>>(partials, Wsc, bsc, af, out);
  dtfd_attn2<<<1, 256, 0, stream>>>(af, Wv2, bv2, Wu2, bu2, Ww2, bw2,
                                    Wbc, bbc, bag_feat, x_concat, out);
  dtfd_copy_rehearsal<<<(BUF_ * D_) / 256, 256, 0, stream>>>(rehearsal, x_concat);
  dtfd_feat_norm<<<NNODES, 256, 0, stream>>>(x_concat, Wf, bf_, xn);
  dtfd_topk<<<NNODES, 256, 0, stream>>>(xn, idx);
  dtfd_gather_h<<<NNODES, 256, 0, stream>>>(x_concat, idx, bag_feat, h);
  dtfd_gcn1<<<NNODES, 256, 0, stream>>>(h, idx, Wg1, Wg1s, bg1, h1);
  dtfd_gcn2<<<B_, 256, 0, stream>>>(h1, idx, Wg2, Wg2s, bg2, out);
}
